// liQGRU_22771916603946
// MI455X (gfx1250) — compile-verified, matrix operand order checked by reading, not run on hardware
//
#include <hip/hip_runtime.h>

typedef __attribute__((ext_vector_type(16))) __bf16 v16bf;
typedef __attribute__((ext_vector_type(2)))  __bf16 v2bf;
typedef __attribute__((ext_vector_type(8)))  float  v8f;
typedef __attribute__((ext_vector_type(4)))  unsigned v4u;
typedef __attribute__((ext_vector_type(8)))  unsigned v8u;

#define WMMA_BF16(a, b, c) \
  __builtin_amdgcn_wmma_f32_16x16x32_bf16(false, (a), false, (b), (short)0, (c), false, false)

// ---------------------------------------------------------------------------
// Swizzled B-fragment layout: for a K x N bf16 matrix, tile (kb, nb) covers
// k in [kb*32, kb*32+32), n in [nb*16, nb*16+16). The 512 bf16 values are
// stored contiguously in per-lane order: lane l holds 16 halves (v16bf),
// element (2r+p) = M[kb*32 + (r>=4?16:0) + (l/16)*8 + (r&3)*2 + p][nb*16 + l%16]
// ---------------------------------------------------------------------------
__device__ inline v16bf load_frag_B_swz(const __bf16* base, int kb, int nb, int ntiles) {
  int l = threadIdx.x & 31;
  const __bf16* p = base + ((size_t)(kb * ntiles + nb) * 512) + (size_t)l * 16;
  return *reinterpret_cast<const v16bf*>(p);
}

// ---------------------------------------------------------------------------
// init: zero barrier counter + ht ping buffer (h0 = 0). 128 blocks x 256.
// ---------------------------------------------------------------------------
__global__ void ligru_init(unsigned* bar, __bf16* ht0, int n) {
  int i = blockIdx.x * 256 + threadIdx.x;
  if (i < n) ht0[i] = (__bf16)0.0f;
  if (i == 0) *bar = 0u;
}

// ---------------------------------------------------------------------------
// Expand quaternion component stack (4, inq, outq) f32 into the dense
// Hamilton block matrix, converted to bf16 and stored in swizzled
// B-fragment layout with Ntot total columns, at column offset n_off.
// ---------------------------------------------------------------------------
__global__ void expand_qweight(const float* __restrict__ w, __bf16* __restrict__ dst,
                               int inq, int outq, int Ntot, int n_off) {
  int N = 4 * outq;
  int K = 4 * inq;
  size_t idx = (size_t)blockIdx.x * 256 + threadIdx.x;
  if (idx >= (size_t)K * N) return;
  int n = (int)(idx % N);
  int k = (int)(idx / N);
  int p = k / inq, a = k % inq;
  int q = n / outq, b = n % outq;
  const int   comp[4][4] = {{0,1,2,3},{1,0,3,2},{2,3,0,1},{3,2,1,0}};
  const float sgn [4][4] = {{ 1.f, 1.f, 1.f, 1.f},
                            {-1.f, 1.f, 1.f,-1.f},
                            {-1.f,-1.f, 1.f, 1.f},
                            {-1.f, 1.f,-1.f, 1.f}};
  float v = sgn[p][q] * w[((size_t)comp[p][q] * inq + a) * outq + b];

  int nn   = n_off + n;
  int kb   = k >> 5, kpos = k & 31;
  int nb   = nn >> 4, mod  = nn & 15;
  int rhi  = (kpos >> 4) & 1;          // k >= 16 within tile -> regs 4..7
  int rem  = kpos & 15;
  int half = rem >> 3;                 // lane half
  int rem8 = rem & 7;
  int r    = rhi * 4 + (rem8 >> 1);
  int pp   = rem8 & 1;
  int lane = half * 16 + mod;
  size_t addr = ((size_t)(kb * (Ntot >> 4) + nb) * 512) + (size_t)lane * 16 + r * 2 + pp;
  dst[addr] = (__bf16)v;
}

// ---------------------------------------------------------------------------
// Input projections: P = x @ [Wh | Wz] + [bh | bz]
// x: (32000 x 1024) f32 row-major; Wswz: swizzled bf16 (K=1024, Ntot=2048).
// One block = 8 waves sharing one 16-row m-tile of x. The m-tile (64 KB f32)
// is DMA'd into LDS once per block by the Tensor Data Mover with pad-on-load
// (2 DWORDs every 256 DWORDs) to skew LDS banks; each wave then builds its
// A-fragments from LDS (f32->bf16) and streams 4 swizzled B-fragments per
// K-step from global, 4 WMMAs per K-step.
// LDS element address for logical dword L = row*1024 + k:  L + ((L>>8)<<1).
// ---------------------------------------------------------------------------
__global__ void __launch_bounds__(256) gemm_in(
    const float* __restrict__ x, const __bf16* __restrict__ Wswz,
    const float* __restrict__ bh, const float* __restrict__ bz,
    float* __restrict__ wh_out, float* __restrict__ wz_out) {
  const int H = 1024;
  __shared__ float xs[16512];           // 16*1024 dwords + 64 pads * 2 dwords

  int mt = blockIdx.x >> 2;                               // m-tile 0..1999
  int ng = ((blockIdx.x & 3) << 3) + (threadIdx.x >> 5);  // n-group 0..31
  int l = threadIdx.x & 31, mod = l & 15, half = l >> 4;

  // ---- TDM: DMA the 16 x 1024 f32 tile of x into LDS (one per block) ----
  if (threadIdx.x == 0) {
    unsigned lds = (unsigned)(uintptr_t)(&xs[0]);   // flat low 32 bits == LDS offset
    unsigned long long ga =
        (unsigned long long)(uintptr_t)x + (unsigned long long)mt * (16ull * 4096ull);
    // group0: count=1 | lds_addr | global_addr[31:0] | addr[56:32] + type=2
    v4u g0 = { 1u, lds, (unsigned)ga, (unsigned)(ga >> 32) | (2u << 30) };
    // group1: data_size=4B, pad_enable, pad_interval=256dw, pad_amount=2dw;
    //         tensor_dim0=1024, tensor_dim1=16, tile_dim0=1024, tile_dim1=16,
    //         tensor_dim0_stride=1024
    v8u g1 = { 0x03D20000u,
               1024u << 16,          // tensor_dim0[15:0] in [31:16]
               16u << 16,            // tensor_dim0[31:16]=0 | tensor_dim1[15:0]
               1024u << 16,          // tensor_dim1[31:16]=0 | tile_dim0
               16u,                  // tile_dim1 | tile_dim2=0
               1024u,                // tensor_dim0_stride[31:0]
               0u, 0u };
    v4u gz = { 0u, 0u, 0u, 0u };     // groups 2/3: tile_dim3/4 = 0 (unused)
    asm volatile("tensor_load_to_lds %0, %1, %2, %3"
                 :: "s"(g0), "s"(g1), "s"(gz), "s"(gz) : "memory");
    __builtin_amdgcn_s_wait_tensorcnt(0);
  }
  __syncthreads();

  int rowbase = mod * 1024;            // per-lane A row (logical dwords)
  int nb0 = ng * 4;
  v8f c0 = {}, c1 = {}, c2 = {}, c3 = {};

  for (int kb = 0; kb < 32; ++kb) {
    v16bf a;
#pragma unroll
    for (int r = 0; r < 8; ++r) {
      int k = kb * 32 + ((r & 4) << 2) + half * 8 + (r & 3) * 2;
      int L = rowbase + k;
      int idx = L + ((L >> 8) << 1);   // pad skew: +2 dwords per 256 dwords
      float2 t = *reinterpret_cast<const float2*>(xs + idx);
      a[2 * r]     = (__bf16)t.x;
      a[2 * r + 1] = (__bf16)t.y;
    }
    v16bf b0 = load_frag_B_swz(Wswz, kb, nb0 + 0, 128);
    v16bf b1 = load_frag_B_swz(Wswz, kb, nb0 + 1, 128);
    v16bf b2 = load_frag_B_swz(Wswz, kb, nb0 + 2, 128);
    v16bf b3 = load_frag_B_swz(Wswz, kb, nb0 + 3, 128);
    c0 = WMMA_BF16(a, b0, c0);
    c1 = WMMA_BF16(a, b1, c1);
    c2 = WMMA_BF16(a, b2, c2);
    c3 = WMMA_BF16(a, b3, c3);
  }

  v8f* cs[4] = {&c0, &c1, &c2, &c3};
#pragma unroll
  for (int i = 0; i < 4; ++i) {
    int n = (nb0 + i) * 16 + mod;                 // 0..2047
    float bias   = (n < H) ? bh[n] : bz[n - H];
    float* dst   = (n < H) ? wh_out : wz_out;
    int col      = n & (H - 1);
    v8f c = *cs[i];
#pragma unroll
    for (int j = 0; j < 8; ++j) {
      int row = mt * 16 + j + 8 * half;
      dst[(size_t)row * H + col] = c[j] + bias;
    }
  }
}

// ---------------------------------------------------------------------------
// Persistent recurrent scan: 16 blocks x 256 threads = 128 waves.
// Wave wid -> (m_tile = wid&1, n_tile = wid>>1). Each wave owns a fixed
// 16x16 tile of h; f32 state lives in registers (C layout) for all T steps.
// Per step: ch = h@Uh tile, cz = h@Uz tile (32 K-iters, 2 WMMA each),
// pointwise gate math, write h to d_out + bf16 double buffer, grid barrier.
// ---------------------------------------------------------------------------
__global__ void __launch_bounds__(256) ligru_scan(
    const float* __restrict__ wh_out, const float* __restrict__ wz_out,
    const __bf16* __restrict__ Uh, const __bf16* __restrict__ Uz,
    const float* __restrict__ mask,
    __bf16* __restrict__ ht0, __bf16* __restrict__ ht1,
    unsigned* __restrict__ bar, float* __restrict__ out) {
  const int T = 1000, B = 32, H = 1024;
  const unsigned NBLK = 16;
  int wid   = (int)((blockIdx.x << 3) + (threadIdx.x >> 5));  // 0..127
  int l     = threadIdx.x & 31, mod = l & 15, half = l >> 4;
  int mtile = wid & 1;
  int ntile = wid >> 1;                 // 0..63
  int arow  = mtile * 16 + mod;         // batch row for A fragment
  int col   = ntile * 16 + mod;         // hidden column for C layout

  float mk[8];
#pragma unroll
  for (int j = 0; j < 8; ++j)
    mk[j] = mask[(size_t)(mtile * 16 + j + 8 * half) * H + col];

  v8f h = {};                           // f32 hidden state, resident in VGPRs
  __bf16* htbuf0 = ht0;
  __bf16* htbuf1 = ht1;

  for (int t = 0; t < T; ++t) {
    const __bf16* hsrc = (t & 1) ? htbuf1 : htbuf0;
    __bf16*       hdst = (t & 1) ? htbuf0 : htbuf1;
    const __bf16* hrow = hsrc + (size_t)arow * H;

    v8f ch = {}, cz = {};
#pragma unroll 2
    for (int kb = 0; kb < 32; ++kb) {
      v16bf a;
#pragma unroll
      for (int r = 0; r < 8; ++r) {
        int k = kb * 32 + ((r & 4) << 2) + half * 8 + (r & 3) * 2;
        v2bf tv = *reinterpret_cast<const v2bf*>(hrow + k);
        a[2 * r]     = tv[0];
        a[2 * r + 1] = tv[1];
      }
      v16bf bh_ = load_frag_B_swz(Uh, kb, ntile, 64);
      v16bf bz_ = load_frag_B_swz(Uz, kb, ntile, 64);
      ch = WMMA_BF16(a, bh_, ch);
      cz = WMMA_BF16(a, bz_, cz);
    }

    const float* whp = wh_out + (size_t)t * B * H;
    const float* wzp = wz_out + (size_t)t * B * H;
    float*       op  = out    + (size_t)t * B * H;
#pragma unroll
    for (int j = 0; j < 8; ++j) {
      int row = mtile * 16 + j + 8 * half;
      size_t off = (size_t)row * H + col;
      float at = whp[off] + ch[j];
      float zp = wzp[off] + cz[j];
      float z  = 1.0f / (1.0f + __expf(-zp));
      float hc = fmaxf(at, 0.0f) * mk[j];
      float hn = z * h[j] + (1.0f - z) * hc;
      h[j] = hn;
      op[off]   = hn;
      hdst[off] = (__bf16)hn;
    }

    // -------- grid-wide barrier (monotonic counter, one per step) --------
    __threadfence();
    __syncthreads();
    if (threadIdx.x == 0) {
      __hip_atomic_fetch_add(bar, 1u, __ATOMIC_RELEASE, __HIP_MEMORY_SCOPE_AGENT);
      unsigned target = (unsigned)(t + 1) * NBLK;
      while (__hip_atomic_load(bar, __ATOMIC_ACQUIRE, __HIP_MEMORY_SCOPE_AGENT) < target)
        __builtin_amdgcn_s_sleep(1);
    }
    __syncthreads();
  }
}

// ---------------------------------------------------------------------------
// Host launcher
// ---------------------------------------------------------------------------
extern "C" void kernel_launch(void* const* d_in, const int* in_sizes, int n_in,
                              void* d_out, int out_size, void* d_ws, size_t ws_size,
                              hipStream_t stream) {
  const float* x    = (const float*)d_in[0];  // (1000, 32, 1024)
  const float* wh_w = (const float*)d_in[1];  // (4, 256, 256)
  const float* wh_b = (const float*)d_in[2];  // (1024,)
  const float* wz_w = (const float*)d_in[3];
  const float* wz_b = (const float*)d_in[4];
  const float* uh_w = (const float*)d_in[5];
  const float* uz_w = (const float*)d_in[6];
  const float* mask = (const float*)d_in[7];  // (32, 1024)

  char* ws = (char*)d_ws;
  // workspace layout (byte offsets, all 256-aligned)
  unsigned* bar   = (unsigned*)(ws + 0);
  __bf16*   ht0   = (__bf16*)(ws + 256);                 // 64 KB
  __bf16*   ht1   = (__bf16*)(ws + 256 + 65536);         // 64 KB
  __bf16*   Wswz  = (__bf16*)(ws + (1u << 20));          // 4 MB  (1024 x 2048 swz)
  __bf16*   UhS   = (__bf16*)(ws + (1u << 20) + (4u << 20));           // 2 MB
  __bf16*   UzS   = (__bf16*)(ws + (1u << 20) + (6u << 20));           // 2 MB
  float*    whout = (float*) (ws + (16u << 20));                        // 131 MB
  float*    wzout = (float*) (ws + (16u << 20) + 131072000u);           // 131 MB

  // 1) zero barrier + h0
  ligru_init<<<128, 256, 0, stream>>>(bar, ht0, 32 * 1024);

  // 2) expand quaternion weights -> swizzled bf16 B-fragment layout
  expand_qweight<<<4096, 256, 0, stream>>>(wh_w, Wswz, 256, 256, 2048, 0);
  expand_qweight<<<4096, 256, 0, stream>>>(wz_w, Wswz, 256, 256, 2048, 1024);
  expand_qweight<<<4096, 256, 0, stream>>>(uh_w, UhS,  256, 256, 1024, 0);
  expand_qweight<<<4096, 256, 0, stream>>>(uz_w, UzS,  256, 256, 1024, 0);

  // 3) big input-projection GEMM (2000 m-tiles x 4 blocks x 8 waves)
  gemm_in<<<8000, 256, 0, stream>>>(x, Wswz, wh_b, wz_b, whout, wzout);

  // 4) persistent recurrent scan
  ligru_scan<<<16, 256, 0, stream>>>(whout, wzout, UhS, UzS, mask,
                                     ht0, ht1, bar, (float*)d_out);
}
